// CLIPEncoderLayer_70738111365679
// MI455X (gfx1250) — compile-verified, matrix operand order checked by reading, not run on hardware
//
#include <hip/hip_runtime.h>

typedef __attribute__((ext_vector_type(16))) __bf16 v16bf;
typedef __attribute__((ext_vector_type(8)))  __bf16 v8bf;
typedef __attribute__((ext_vector_type(8)))  float  v8f;

#define E_DIM 768
#define H_NUM 12
#define D_DIM 64
#define F_DIM 3072
#define S_LEN 1024
#define B_NUM 8
#define N_TOK (B_NUM * S_LEN)   // 8192 tokens

// ---------------------------------------------------------------------------
// WMMA helpers
// ---------------------------------------------------------------------------
__device__ __forceinline__ v8f zero8() {
  v8f z;
#pragma unroll
  for (int i = 0; i < 8; ++i) z[i] = 0.0f;
  return z;
}

__device__ __forceinline__ v8f wmma_bf16(v16bf a, v16bf b, v8f c) {
  // D = A(16x32 bf16) * B(32x16 bf16) + C(16x16 f32)
  return __builtin_amdgcn_wmma_f32_16x16x32_bf16(
      /*neg_a=*/false, a, /*neg_b=*/false, b,
      /*c_mod=*/(short)0, c, /*reuse_a=*/false, /*reuse_b=*/false);
}

// Load a 16x32 bf16 fragment in the CDNA5 16-bit A layout from a row-major
// tile (row stride `ld` elements, rows 16B aligned).  Lane m=lane&15 carries
// row M=m; elements 0..7 = K hi*8+0..7, elements 8..15 = K 16+hi*8+0..7.
// The same layout serves B fragments when the B tile is stored transposed
// (Bt[n][k]).
__device__ __forceinline__ v16bf load_frag(const __bf16* __restrict__ base, int ld) {
  const int lane = threadIdx.x & 31;
  const int r  = lane & 15;
  const int hi = lane >> 4;
  const __bf16* p = base + r * ld + hi * 8;
  v8bf a0 = *reinterpret_cast<const v8bf*>(p);
  v8bf a1 = *reinterpret_cast<const v8bf*>(p + 16);
  v16bf f;
#pragma unroll
  for (int i = 0; i < 8; ++i) { f[i] = a0[i]; f[i + 8] = a1[i]; }
  return f;
}

// ---------------------------------------------------------------------------
// Elementwise conversion / packing kernels
// ---------------------------------------------------------------------------
__global__ __launch_bounds__(256) void convert_bf16_kernel(
    const float* __restrict__ in, __bf16* __restrict__ out, int n) {
  int i = blockIdx.x * 256 + threadIdx.x;
  if (i < n) out[i] = (__bf16)in[i];
}

__global__ __launch_bounds__(256) void pack_qkv_w_kernel(
    const float* __restrict__ wq, const float* __restrict__ wk,
    const float* __restrict__ wv, __bf16* __restrict__ out) {
  int i = blockIdx.x * 256 + threadIdx.x;
  if (i >= E_DIM * 3 * E_DIM) return;
  int k = i / (3 * E_DIM);
  int c = i % (3 * E_DIM);
  float v;
  if (c < E_DIM)            v = wq[k * E_DIM + c];
  else if (c < 2 * E_DIM)   v = wk[k * E_DIM + (c - E_DIM)];
  else                      v = wv[k * E_DIM + (c - 2 * E_DIM)];
  out[i] = (__bf16)v;
}

__global__ __launch_bounds__(256) void pack_qkv_b_kernel(
    const float* __restrict__ bq, const float* __restrict__ bk,
    const float* __restrict__ bv, float* __restrict__ out) {
  int c = blockIdx.x * 256 + threadIdx.x;
  if (c >= 3 * E_DIM) return;
  out[c] = (c < E_DIM) ? bq[c] : (c < 2 * E_DIM) ? bk[c - E_DIM] : bv[c - 2 * E_DIM];
}

// ---------------------------------------------------------------------------
// LayerNorm (one block of 256 threads per token; E=768 -> 3 elems/thread),
// writes bf16 normalized activations.
// ---------------------------------------------------------------------------
__global__ __launch_bounds__(256) void layernorm_bf16_kernel(
    const float* __restrict__ x, const float* __restrict__ gamma,
    const float* __restrict__ beta, __bf16* __restrict__ out) {
  __shared__ float red[16];
  const int row = blockIdx.x;
  const int t = threadIdx.x;
  const float* xr = x + (size_t)row * E_DIM;
  float v0 = xr[t], v1 = xr[t + 256], v2 = xr[t + 512];
  float s  = v0 + v1 + v2;
  float s2 = v0 * v0 + v1 * v1 + v2 * v2;
#pragma unroll
  for (int o = 16; o > 0; o >>= 1) {
    s  += __shfl_xor(s,  o, 32);
    s2 += __shfl_xor(s2, o, 32);
  }
  if ((t & 31) == 0) { red[t >> 5] = s; red[8 + (t >> 5)] = s2; }
  __syncthreads();
  float S = 0.0f, S2 = 0.0f;
#pragma unroll
  for (int i = 0; i < 8; ++i) { S += red[i]; S2 += red[8 + i]; }
  const float mu  = S * (1.0f / E_DIM);
  const float var = S2 * (1.0f / E_DIM) - mu * mu;
  const float rs  = rsqrtf(var + 1e-5f);
  __bf16* orow = out + (size_t)row * E_DIM;
  orow[t]       = (__bf16)((v0 - mu) * rs * gamma[t]       + beta[t]);
  orow[t + 256] = (__bf16)((v1 - mu) * rs * gamma[t + 256] + beta[t + 256]);
  orow[t + 512] = (__bf16)((v2 - mu) * rs * gamma[t + 512] + beta[t + 512]);
}

// ---------------------------------------------------------------------------
// Generic WMMA GEMM:  out(N x M) = A(N x K, bf16) @ W(K x M, bf16) + bias.
// 128x128 block tile, K-step 32, 8 waves (4 row-groups x 2 col-groups), each
// wave computes a 32x64 sub-tile = 8 WMMAs per K-step.
// Double-buffered LDS: next tile's global loads issue before this tile's
// WMMAs, register-staged, committed to the other buffer, one barrier/step.
// ---------------------------------------------------------------------------
enum { EPI_QKV = 0, EPI_RES = 1, EPI_GELU = 2 };

template <int EPI>
__global__ __launch_bounds__(256) void gemm_bf16_kernel(
    const __bf16* __restrict__ A, const __bf16* __restrict__ W,
    const float* __restrict__ bias, const float* __restrict__ resid,
    float* __restrict__ outF, __bf16* __restrict__ outB,
    __bf16* __restrict__ qO, __bf16* __restrict__ kO, __bf16* __restrict__ vO,
    int K, int M) {
  __shared__ __bf16 As[2][128][40];  // [buf][row][k]
  __shared__ __bf16 Bs[2][128][40];  // [buf][col][k]  (transposed W tile)
  const int tid = threadIdx.x;
  const int wid = tid >> 5;
  const int wm = wid & 3;    // wave row group: 32 rows each
  const int wn = wid >> 2;   // wave col group: 64 cols each
  const int row0 = blockIdx.y * 128;
  const int col0 = blockIdx.x * 128;

  v8f acc[2][4];
#pragma unroll
  for (int i = 0; i < 2; ++i)
#pragma unroll
    for (int j = 0; j < 4; ++j) acc[i][j] = zero8();

  const int arow = tid >> 1, aseg = (tid & 1) << 4;   // A tile: 128 rows x 32 k
  const int wk_ = tid >> 3, wc = (tid & 7) << 4;      // W tile: 32 k x 128 cols
  const __bf16* gA = A + (size_t)(row0 + arow) * K + aseg;
  const __bf16* gW = W + (size_t)wk_ * M + (col0 + wc);

  // Stage tile 0.
  {
    v8bf a0 = *reinterpret_cast<const v8bf*>(gA);
    v8bf a1 = *reinterpret_cast<const v8bf*>(gA + 8);
    *reinterpret_cast<v8bf*>(&As[0][arow][aseg])     = a0;
    *reinterpret_cast<v8bf*>(&As[0][arow][aseg + 8]) = a1;
    v8bf b0 = *reinterpret_cast<const v8bf*>(gW);
    v8bf b1 = *reinterpret_cast<const v8bf*>(gW + 8);
#pragma unroll
    for (int i = 0; i < 8; ++i) {
      Bs[0][wc + i][wk_]     = b0[i];
      Bs[0][wc + 8 + i][wk_] = b1[i];
    }
  }
  __syncthreads();

  const int steps = K >> 5;
  for (int it = 0; it < steps; ++it) {
    const int cur = it & 1;
    const bool have = (it + 1) < steps;
    v8bf na0{}, na1{}, nb0{}, nb1{};
    if (have) {  // issue next tile's global loads before this tile's math
      const __bf16* ga2 = gA + (size_t)(it + 1) * 32;
      na0 = *reinterpret_cast<const v8bf*>(ga2);
      na1 = *reinterpret_cast<const v8bf*>(ga2 + 8);
      const __bf16* gw2 = gW + (size_t)(it + 1) * 32 * M;
      nb0 = *reinterpret_cast<const v8bf*>(gw2);
      nb1 = *reinterpret_cast<const v8bf*>(gw2 + 8);
    }

    v16bf af0 = load_frag(&As[cur][wm * 32][0], 40);
    v16bf af1 = load_frag(&As[cur][wm * 32 + 16][0], 40);
#pragma unroll
    for (int j = 0; j < 4; ++j) {
      v16bf bfj = load_frag(&Bs[cur][wn * 64 + j * 16][0], 40);
      acc[0][j] = wmma_bf16(af0, bfj, acc[0][j]);
      acc[1][j] = wmma_bf16(af1, bfj, acc[1][j]);
    }

    if (have) {  // commit next tile to the other buffer
      const int nxt = cur ^ 1;
      *reinterpret_cast<v8bf*>(&As[nxt][arow][aseg])     = na0;
      *reinterpret_cast<v8bf*>(&As[nxt][arow][aseg + 8]) = na1;
#pragma unroll
      for (int i = 0; i < 8; ++i) {
        Bs[nxt][wc + i][wk_]     = nb0[i];
        Bs[nxt][wc + 8 + i][wk_] = nb1[i];
      }
    }
    __syncthreads();
  }

  // Epilogue. C layout: element r of v8f -> row hi*8+r, lane&15 -> col.
  const int lane = tid & 31, nn = lane & 15, hi = lane >> 4;
#pragma unroll
  for (int i = 0; i < 2; ++i)
#pragma unroll
    for (int j = 0; j < 4; ++j)
#pragma unroll
      for (int r = 0; r < 8; ++r) {
        const int row = row0 + wm * 32 + i * 16 + hi * 8 + r;
        const int col = col0 + wn * 64 + j * 16 + nn;
        float v = acc[i][j][r] + bias[col];
        if (EPI == EPI_QKV) {
          const int which = col / E_DIM;
          const int e = col % E_DIM;
          const int h = e >> 6, d = e & 63;
          const int b = row >> 10, s = row & 1023;
          const size_t o = ((size_t)(b * H_NUM + h) * S_LEN + s) * D_DIM + d;
          if (which == 0)      qO[o] = (__bf16)(v * 0.125f);  // * D^-0.5
          else if (which == 1) kO[o] = (__bf16)v;
          else                 vO[o] = (__bf16)v;
        } else if (EPI == EPI_RES) {
          const size_t o = (size_t)row * M + col;
          outF[o] = v + resid[o];
        } else {  // EPI_GELU (exact)
          const float g = 0.5f * v * (1.0f + erff(v * 0.70710678118f));
          outB[(size_t)row * M + col] = (__bf16)g;
        }
      }
}

// ---------------------------------------------------------------------------
// Online-softmax update over a 16x64 score strip.  MASK specialization is
// selected by a wave-uniform branch so off-diagonal tiles pay no v_cndmask.
// ---------------------------------------------------------------------------
template <bool MASK>
__device__ __forceinline__ void softmax_update(
    v8f (&sf)[4], v8f (&oacc)[4], float (&mrun)[8], float (&lrun)[8],
    int rowbase, int kv0, int nn, int hi) {
#pragma unroll
  for (int r = 0; r < 8; ++r) {
    const int rowg = rowbase + hi * 8 + r;
    float mx = -1e30f;
#pragma unroll
    for (int j = 0; j < 4; ++j) {
      float sv = sf[j][r];
      if (MASK) {
        const int colg = kv0 + j * 16 + nn;
        if (colg > rowg) sv = -1e30f;
        sf[j][r] = sv;
      }
      mx = fmaxf(mx, sv);
    }
#pragma unroll
    for (int o = 8; o > 0; o >>= 1) mx = fmaxf(mx, __shfl_xor(mx, o, 16));
    const float mnew = fmaxf(mrun[r], mx);
    const float fac = __expf(mrun[r] - mnew);
#pragma unroll
    for (int c = 0; c < 4; ++c) oacc[c][r] *= fac;
    float ps = 0.0f;
#pragma unroll
    for (int j = 0; j < 4; ++j) {
      const float p = __expf(sf[j][r] - mnew);
      sf[j][r] = p;
      ps += p;
    }
#pragma unroll
    for (int o = 8; o > 0; o >>= 1) ps += __shfl_xor(ps, o, 16);
    lrun[r] = lrun[r] * fac + ps;
    mrun[r] = mnew;
  }
}

// ---------------------------------------------------------------------------
// Flash attention, causal.  One block = one (b,h) and 128 query rows; 8 waves,
// each wave owns 16 query rows.  kv tiles of 64 are double-buffered through
// LDS; the analytic causal mask stops the kv loop at the diagonal and waves
// skip tiles entirely above their rows.
// ---------------------------------------------------------------------------
__global__ __launch_bounds__(256) void attention_kernel(
    const __bf16* __restrict__ Q, const __bf16* __restrict__ Kb,
    const __bf16* __restrict__ Vb, __bf16* __restrict__ O) {
  __shared__ __bf16 Ks[2][64][72];   // [buf][kv][d]  (natural K layout == Bt for Q@K^T)
  __shared__ __bf16 Vt[2][64][72];   // [buf][d][kv]  (transposed V == Bt for P@V)
  __shared__ __bf16 Ps[8][16][72];   // per-wave P tile (C-layout -> A-layout bounce)

  const int bh = blockIdx.x;               // b*H + h
  const int q0 = blockIdx.y * 128;
  const int tid = threadIdx.x;
  const int wid = tid >> 5;
  const int lane = tid & 31, nn = lane & 15, hi = lane >> 4;
  const size_t base = (size_t)bh * (S_LEN * D_DIM);
  const int wrow0 = q0 + wid * 16;         // this wave's first query row

  // Persistent Q fragments straight from global (rows wid*16.., D=64 -> 2 K-steps).
  const __bf16* qp = Q + base + (size_t)wrow0 * D_DIM;
  const v16bf qf0 = load_frag(qp, D_DIM);
  const v16bf qf1 = load_frag(qp + 32, D_DIM);

  v8f oacc[4];
#pragma unroll
  for (int c = 0; c < 4; ++c) oacc[c] = zero8();
  float mrun[8], lrun[8];
#pragma unroll
  for (int r = 0; r < 8; ++r) { mrun[r] = -1e30f; lrun[r] = 0.0f; }

  const int krow = tid >> 2, kseg = (tid & 3) << 4;  // 64x64 tile, 16 elems/thread
  const __bf16* gK = Kb + base + (size_t)krow * D_DIM + kseg;
  const __bf16* gV = Vb + base + (size_t)krow * D_DIM + kseg;

  // Stage kv tile 0.
  {
    v8bf k0v = *reinterpret_cast<const v8bf*>(gK);
    v8bf k1v = *reinterpret_cast<const v8bf*>(gK + 8);
    *reinterpret_cast<v8bf*>(&Ks[0][krow][kseg])     = k0v;
    *reinterpret_cast<v8bf*>(&Ks[0][krow][kseg + 8]) = k1v;
    v8bf v0v = *reinterpret_cast<const v8bf*>(gV);
    v8bf v1v = *reinterpret_cast<const v8bf*>(gV + 8);
#pragma unroll
    for (int i = 0; i < 8; ++i) {
      Vt[0][kseg + i][krow]     = v0v[i];
      Vt[0][kseg + 8 + i][krow] = v1v[i];
    }
  }
  __syncthreads();

  const int steps = (q0 + 128) >> 6;
  for (int it = 0; it < steps; ++it) {
    const int cur = it & 1;
    const int kv0 = it << 6;
    const bool have = (it + 1) < steps;
    v8bf nk0{}, nk1{}, nv0{}, nv1{};
    if (have) {  // issue next tile's loads before this tile's math
      const __bf16* gk2 = gK + (size_t)(it + 1) * 64 * D_DIM;
      nk0 = *reinterpret_cast<const v8bf*>(gk2);
      nk1 = *reinterpret_cast<const v8bf*>(gk2 + 8);
      const __bf16* gv2 = gV + (size_t)(it + 1) * 64 * D_DIM;
      nv0 = *reinterpret_cast<const v8bf*>(gv2);
      nv1 = *reinterpret_cast<const v8bf*>(gv2 + 8);
    }

    if (kv0 <= wrow0 + 15) {  // wave-uniform: tile not entirely above diagonal
      // Scores: 16 q-rows x 64 kv-cols = 4 frags, 2 K-steps over D.
      v8f sf[4];
#pragma unroll
      for (int j = 0; j < 4; ++j) {
        v16bf kf0 = load_frag(&Ks[cur][j * 16][0], 72);
        v16bf kf1 = load_frag(&Ks[cur][j * 16][32], 72);
        sf[j] = wmma_bf16(qf0, kf0, zero8());
        sf[j] = wmma_bf16(qf1, kf1, sf[j]);
      }

      if (kv0 + 63 > wrow0)  // diagonal tile: needs per-element masking
        softmax_update<true>(sf, oacc, mrun, lrun, wrow0, kv0, nn, hi);
      else                   // strictly below diagonal: mask-free
        softmax_update<false>(sf, oacc, mrun, lrun, wrow0, kv0, nn, hi);

      // C-layout f32 -> A-layout bf16 via wave-private LDS.
#pragma unroll
      for (int j = 0; j < 4; ++j)
#pragma unroll
        for (int r = 0; r < 8; ++r)
          Ps[wid][hi * 8 + r][j * 16 + nn] = (__bf16)sf[j][r];

      // O += P @ V   (2 K-steps over kv, 4 d-frags)
#pragma unroll
      for (int kk = 0; kk < 2; ++kk) {
        v16bf paf = load_frag(&Ps[wid][0][kk * 32], 72);
#pragma unroll
        for (int c = 0; c < 4; ++c) {
          v16bf vf = load_frag(&Vt[cur][c * 16][kk * 32], 72);
          oacc[c] = wmma_bf16(paf, vf, oacc[c]);
        }
      }
    }

    if (have) {  // commit next kv tile into the other buffer
      const int nxt = cur ^ 1;
      *reinterpret_cast<v8bf*>(&Ks[nxt][krow][kseg])     = nk0;
      *reinterpret_cast<v8bf*>(&Ks[nxt][krow][kseg + 8]) = nk1;
#pragma unroll
      for (int i = 0; i < 8; ++i) {
        Vt[nxt][kseg + i][krow]     = nv0[i];
        Vt[nxt][kseg + 8 + i][krow] = nv1[i];
      }
    }
    __syncthreads();
  }

  // Normalize and scatter to token-major (B,S,H*D) bf16 for the O-projection.
  const int b = bh / H_NUM, h = bh % H_NUM;
#pragma unroll
  for (int c = 0; c < 4; ++c)
#pragma unroll
    for (int r = 0; r < 8; ++r) {
      const int s = q0 + wid * 16 + hi * 8 + r;
      const float val = oacc[c][r] / lrun[r];
      O[(size_t)(b * S_LEN + s) * E_DIM + h * D_DIM + c * 16 + nn] = (__bf16)val;
    }
}

// ---------------------------------------------------------------------------
// Host launcher
// ---------------------------------------------------------------------------
extern "C" void kernel_launch(void* const* d_in, const int* in_sizes, int n_in,
                              void* d_out, int out_size, void* d_ws, size_t ws_size,
                              hipStream_t stream) {
  (void)in_sizes; (void)n_in; (void)out_size; (void)ws_size;
  const float* x     = (const float*)d_in[0];
  /* d_in[1] = mask: causal, applied analytically, never read */
  const float* ln1_g = (const float*)d_in[2];
  const float* ln1_b = (const float*)d_in[3];
  const float* wq    = (const float*)d_in[4];
  const float* bq    = (const float*)d_in[5];
  const float* wk    = (const float*)d_in[6];
  const float* bk    = (const float*)d_in[7];
  const float* wv    = (const float*)d_in[8];
  const float* bv    = (const float*)d_in[9];
  const float* wo    = (const float*)d_in[10];
  const float* bo    = (const float*)d_in[11];
  const float* ln2_g = (const float*)d_in[12];
  const float* ln2_b = (const float*)d_in[13];
  const float* w1    = (const float*)d_in[14];
  const float* b1    = (const float*)d_in[15];
  const float* w2    = (const float*)d_in[16];
  const float* b2    = (const float*)d_in[17];
  float* outp = (float*)d_out;

  // Workspace carving (256B aligned).
  char* ws = (char*)d_ws;
  auto carve = [&](size_t bytes) -> void* {
    void* p = ws;
    ws += (bytes + 255) & ~(size_t)255;
    return p;
  };
  __bf16* wqkvB = (__bf16*)carve((size_t)E_DIM * 3 * E_DIM * 2);
  float*  bqkvF = (float*) carve((size_t)3 * E_DIM * 4);
  __bf16* woB   = (__bf16*)carve((size_t)E_DIM * E_DIM * 2);
  __bf16* w1B   = (__bf16*)carve((size_t)E_DIM * F_DIM * 2);
  __bf16* w2B   = (__bf16*)carve((size_t)F_DIM * E_DIM * 2);
  __bf16* h1B   = (__bf16*)carve((size_t)N_TOK * E_DIM * 2);  // LN1 output
  __bf16* qB    = (__bf16*)carve((size_t)N_TOK * E_DIM * 2);  // (B,H,S,D)
  __bf16* kB    = (__bf16*)carve((size_t)N_TOK * E_DIM * 2);
  __bf16* vB    = (__bf16*)carve((size_t)N_TOK * E_DIM * 2);
  __bf16* attnB = (__bf16*)carve((size_t)N_TOK * E_DIM * 2);  // (B,S,E)
  float*  hresF = (float*) carve((size_t)N_TOK * E_DIM * 4);  // x + attn@wo
  __bf16* m2B   = (__bf16*)carve((size_t)N_TOK * E_DIM * 2);  // LN2 output
  __bf16* ff1B  = (__bf16*)carve((size_t)N_TOK * F_DIM * 2);  // gelu(m@w1+b1)

  // --- weight conversion / packing -----------------------------------------
  pack_qkv_w_kernel<<<(E_DIM * 3 * E_DIM + 255) / 256, 256, 0, stream>>>(wq, wk, wv, wqkvB);
  pack_qkv_b_kernel<<<(3 * E_DIM + 255) / 256, 256, 0, stream>>>(bq, bk, bv, bqkvF);
  convert_bf16_kernel<<<(E_DIM * E_DIM + 255) / 256, 256, 0, stream>>>(wo, woB, E_DIM * E_DIM);
  convert_bf16_kernel<<<(E_DIM * F_DIM + 255) / 256, 256, 0, stream>>>(w1, w1B, E_DIM * F_DIM);
  convert_bf16_kernel<<<(F_DIM * E_DIM + 255) / 256, 256, 0, stream>>>(w2, w2B, F_DIM * E_DIM);

  // --- LN1 + fused QKV projection ------------------------------------------
  layernorm_bf16_kernel<<<N_TOK, 256, 0, stream>>>(x, ln1_g, ln1_b, h1B);
  gemm_bf16_kernel<EPI_QKV><<<dim3(3 * E_DIM / 128, N_TOK / 128), 256, 0, stream>>>(
      h1B, wqkvB, bqkvF, nullptr, nullptr, nullptr, qB, kB, vB, E_DIM, 3 * E_DIM);

  // --- causal flash attention ----------------------------------------------
  attention_kernel<<<dim3(B_NUM * H_NUM, S_LEN / 128), 256, 0, stream>>>(qB, kB, vB, attnB);

  // --- O projection + residual ---------------------------------------------
  gemm_bf16_kernel<EPI_RES><<<dim3(E_DIM / 128, N_TOK / 128), 256, 0, stream>>>(
      attnB, woB, bo, x, hresF, nullptr, nullptr, nullptr, nullptr, E_DIM, E_DIM);

  // --- LN2 + MLP ------------------------------------------------------------
  layernorm_bf16_kernel<<<N_TOK, 256, 0, stream>>>(hresF, ln2_g, ln2_b, m2B);
  gemm_bf16_kernel<EPI_GELU><<<dim3(F_DIM / 128, N_TOK / 128), 256, 0, stream>>>(
      m2B, w1B, b1, nullptr, nullptr, ff1B, nullptr, nullptr, nullptr, E_DIM, F_DIM);
  gemm_bf16_kernel<EPI_RES><<<dim3(E_DIM / 128, N_TOK / 128), 256, 0, stream>>>(
      ff1B, w2B, b2, hresF, outp, nullptr, nullptr, nullptr, nullptr, F_DIM, E_DIM);
}
